// W8X8Linear_41850161332881
// MI455X (gfx1250) — 8-bit matrix operand order repaired and checked by reading, compiled, not run on hardware
//
#include <hip/hip_runtime.h>
#include <hip/hip_fp16.h>
#include <stdint.h>

typedef __attribute__((ext_vector_type(8))) int          v8i;
typedef __attribute__((ext_vector_type(4))) int          v4i;
typedef __attribute__((ext_vector_type(2))) int          v2i;
typedef __attribute__((ext_vector_type(4))) unsigned int v4u;

#define K_DIM 1024
#define N_DIM 1024

#define KC         256
#define NCHUNK     (K_DIM / KC)
#define ROW_STRIDE (KC + 16)
#define MAT_BYTES  (128 * ROW_STRIDE)
#define BUF_BYTES  (2 * MAT_BYTES)
#define LDS_BYTES  (2 * BUF_BYTES)

__global__ __launch_bounds__(256) void quant_rows(const float* __restrict__ src,
                                                  int8_t* __restrict__ dstq,
                                                  float* __restrict__ scales) {
    const int row = blockIdx.x;
    const int t   = threadIdx.x;

    const float4* s4 = (const float4*)(src + (size_t)row * K_DIM);
    float4 v = s4[t];

    float m = fmaxf(fmaxf(fabsf(v.x), fabsf(v.y)), fmaxf(fabsf(v.z), fabsf(v.w)));

    __shared__ float red[256];
    red[t] = m;
    __syncthreads();
    for (int s = 128; s > 0; s >>= 1) {
        if (t < s) red[t] = fmaxf(red[t], red[t + s]);
        __syncthreads();
    }
    const float mx    = red[0];
    const float scale = 127.0f / mx;

    char4 q;
    q.x = (signed char)__builtin_rintf(v.x * scale);
    q.y = (signed char)__builtin_rintf(v.y * scale);
    q.z = (signed char)__builtin_rintf(v.z * scale);
    q.w = (signed char)__builtin_rintf(v.w * scale);
    ((char4*)(dstq + (size_t)row * K_DIM))[t] = q;

    if (t == 0) scales[row] = mx;
}

__device__ __forceinline__ void tdm_load_tile(uint64_t gaddr, uint32_t lds_off) {
    v4u g0;
    g0[0] = 1u;
    g0[1] = lds_off;
    g0[2] = (uint32_t)(gaddr & 0xFFFFFFFFu);
    g0[3] = (uint32_t)((gaddr >> 32) & 0x01FFFFFFu)
            | 0x80000000u;

    v8i g1;
    g1[0] = (1 << 20)
          | (5 << 22)
          | (3 << 25);
    g1[1] = (int)((K_DIM & 0xFFFF) << 16);
    g1[2] = (int)((K_DIM >> 16) & 0xFFFF)
          | (int)(128u << 16);
    g1[3] = (int)(KC << 16);
    g1[4] = 128;
    g1[5] = K_DIM;
    g1[6] = 0;
    g1[7] = 0;

    v4i g2 = {0, 0, 0, 0};
    v4i g3 = {0, 0, 0, 0};
    v8i g4 = {0, 0, 0, 0, 0, 0, 0, 0};

    __builtin_amdgcn_tensor_load_to_lds(g0, g1, g2, g3, g4, 0);
}

__global__ __launch_bounds__(256) void gemm_w8a8(const int8_t* __restrict__ xq,
                                                 const int8_t* __restrict__ wq,
                                                 const float* __restrict__ xmax,
                                                 const float* __restrict__ wmax,
                                                 const float* __restrict__ bias,
                                                 float* __restrict__ out) {
    extern __shared__ int8_t smem[];

    const int tid   = threadIdx.x;
    const int lane  = tid & 31;
    const int wave  = tid >> 5;
    const int waveM = wave >> 2;
    const int waveN = wave & 3;

    const int mBase = blockIdx.y * 128 + waveM * 64;
    const int nBase = blockIdx.x * 128 + waveN * 32;

    const int c = lane & 15;
    const int h = lane >> 4;

    const uint64_t aG = (uint64_t)(uintptr_t)xq + (uint64_t)blockIdx.y * 128u * K_DIM;
    const uint64_t bG = (uint64_t)(uintptr_t)wq + (uint64_t)blockIdx.x * 128u * K_DIM;

    v8i acc[4][2];
#pragma unroll
    for (int mt = 0; mt < 4; ++mt)
#pragma unroll
        for (int nt = 0; nt < 2; ++nt)
#pragma unroll
            for (int r = 0; r < 8; ++r) acc[mt][nt][r] = 0;

    if (wave == 0) {
        tdm_load_tile(aG, 0u);
        tdm_load_tile(bG, (uint32_t)MAT_BYTES);
    }

    for (int kc = 0; kc < NCHUNK; ++kc) {
        const int buf = kc & 1;

        if (wave == 0) {
            if (kc + 1 < NCHUNK) {
                const int      nbuf = buf ^ 1;
                const uint64_t koff = (uint64_t)(kc + 1) * KC;
                tdm_load_tile(aG + koff, (uint32_t)(nbuf * BUF_BYTES));
                tdm_load_tile(bG + koff, (uint32_t)(nbuf * BUF_BYTES + MAT_BYTES));
                __builtin_amdgcn_s_wait_tensorcnt(2);
            } else {
                __builtin_amdgcn_s_wait_tensorcnt(0);
            }
        }
        __syncthreads();

        const int8_t* aBuf = smem + buf * BUF_BYTES;
        const int8_t* bBuf = smem + buf * BUF_BYTES + MAT_BYTES;

#pragma unroll
        for (int k = 0; k < KC; k += 64) {
            v8i a[4];
#pragma unroll
            for (int mt = 0; mt < 4; ++mt) {
                const int arow = (waveM * 64 + mt * 16 + c) * ROW_STRIDE + k;
#pragma unroll
                for (int p = 0; p < 4; ++p) {
                    v2i t2 = *(const v2i*)(aBuf + arow + (p >> 1) * 32 + h * 16 + (p & 1) * 8);
                    a[mt][2 * p]     = t2[0];
                    a[mt][2 * p + 1] = t2[1];
                }
            }
            v8i b[2];
#pragma unroll
            for (int nt = 0; nt < 2; ++nt) {
                const int brow = (waveN * 32 + nt * 16 + c) * ROW_STRIDE + k + h * 16;
                v4i lo = *(const v4i*)(bBuf + brow);
                v4i hi = *(const v4i*)(bBuf + brow + 32);
#pragma unroll
                for (int r = 0; r < 4; ++r) {
                    b[nt][r]     = lo[r];
                    b[nt][r + 4] = hi[r];
                }
            }
#pragma unroll
            for (int mt = 0; mt < 4; ++mt)
#pragma unroll
                for (int nt = 0; nt < 2; ++nt)
                    acc[mt][nt] = __builtin_amdgcn_wmma_i32_16x16x64_iu8(
                        true, a[mt], true, b[nt], acc[mt][nt], false, false);
        }
        __syncthreads();
    }

    const float inv = 1.0f / (127.0f * 127.0f);
#pragma unroll
    for (int nt = 0; nt < 2; ++nt) {
        const int   n  = nBase + nt * 16 + c;
        const float wm = wmax[n];
        const float bv = bias[n];
#pragma unroll
        for (int mt = 0; mt < 4; ++mt) {
#pragma unroll
            for (int r = 0; r < 8; ++r) {
                const int   m = mBase + mt * 16 + h * 8 + r;
                const float q = (float)acc[mt][nt][r];
                const __half hq = __float2half(q * inv);
                out[(size_t)m * N_DIM + n] = __half2float(hq) * (xmax[m] * wm) + bv;
            }
        }
    }
}

extern "C" void kernel_launch(void* const* d_in, const int* in_sizes, int n_in,
                              void* d_out, int out_size, void* d_ws, size_t ws_size,
                              hipStream_t stream) {
    const float* x    = (const float*)d_in[0];
    const float* w    = (const float*)d_in[1];
    const float* bias = (const float*)d_in[2];
    float* out        = (float*)d_out;

    const int M = in_sizes[0] / K_DIM;

    uint8_t* ws   = (uint8_t*)d_ws;
    int8_t*  wq   = (int8_t*)ws;
    int8_t*  xq   = (int8_t*)(ws + (size_t)N_DIM * K_DIM);
    float*   wmax = (float*)(ws + (size_t)N_DIM * K_DIM + (size_t)M * K_DIM);
    float*   xmax = wmax + N_DIM;

    quant_rows<<<N_DIM, 256, 0, stream>>>(w, wq, wmax);
    quant_rows<<<M,     256, 0, stream>>>(x, xq, xmax);

    dim3 grid(N_DIM / 128, M / 128);
    gemm_w8a8<<<grid, 256, LDS_BYTES, stream>>>(xq, wq, xmax, wmax, bias, out);
}
